// Music2VecModel_30099130810690
// MI455X (gfx1250) — matrix operand order repaired and checked by reading, not compiled, not run on hardware
//
#include <hip/hip_runtime.h>
#include <hip/hip_bf16.h>

// ---------------------------------------------------------------------------
// Music2Vec encoder forward for MI455X (gfx1250, wave32, WMMA).
// All GEMMs + attention run through v_wmma_f32_16x16x32_bf16 with fp32
// accumulation; GEMM K-loop is software-pipelined (global prefetch of the
// next tile overlaps WMMA compute of the current tile).
// ---------------------------------------------------------------------------

typedef __attribute__((ext_vector_type(16))) __bf16 v16bf;
typedef __attribute__((ext_vector_type(8)))  __bf16 v8bf;
typedef __attribute__((ext_vector_type(8)))  float  v8f;

#define NB     2
#define SEQ    2048
#define DIM    768
#define NH     12
#define HDIM   64
#define FF     3072
#define NLAYER 12
#define KW     128
#define CPG    48            // DIM / groups(16)
#define MROWS  (NB * SEQ)    // 4096
#define ATT_SCALE 0.125f     // HDIM^-0.5

__device__ __forceinline__ float geluf(float x) {
  return 0.5f * x * (1.0f + erff(x * 0.70710678118654752f));
}

__device__ __forceinline__ v16bf cat16(v8bf lo, v8bf hi) {
  return __builtin_shufflevector(lo, hi, 0, 1, 2, 3, 4, 5, 6, 7,
                                         8, 9, 10, 11, 12, 13, 14, 15);
}

__device__ __forceinline__ v8f wmma_bf(v16bf a, v16bf b, v8f c) {
  // D = A(16x32 bf16) * B(32x16 bf16) + C(16x16 f32)
  return __builtin_amdgcn_wmma_f32_16x16x32_bf16(false, a, false, b,
                                                 (short)0, c, false, false);
}

// ---------------------------------------------------------------------------
// fp32 -> bf16 elementwise convert (weights)
// ---------------------------------------------------------------------------
__global__ __launch_bounds__(256) void cvt_bf16_kernel(
    const float* __restrict__ src, __bf16* __restrict__ dst, int n) {
  int i = blockIdx.x * 256 + threadIdx.x;
  if (i < n) dst[i] = (__bf16)src[i];
}

// ---------------------------------------------------------------------------
// Grouped positional conv (D=768, 16 groups, K=128, pad 64/64, drop last
// frame) + bias + exact GELU + residual:  out = x + gelu(conv(x)+b)
// ---------------------------------------------------------------------------
__global__ __launch_bounds__(256) void conv_pos_kernel(
    const float* __restrict__ x, const float* __restrict__ w,
    const float* __restrict__ cb, float* __restrict__ out) {
  int idx = blockIdx.x * 256 + threadIdx.x;       // over B*L*D
  int d = idx % DIM;
  int l = (idx / DIM) % SEQ;
  int b = idx / (DIM * SEQ);
  int g = d / CPG;
  const float* wd = w + (size_t)d * CPG * KW;     // (d, c, t)
  const float* xb = x + (size_t)b * SEQ * DIM + g * CPG;

  int t0 = 64 - l;        if (t0 < 0) t0 = 0;
  int t1 = SEQ + 64 - l;  if (t1 > KW) t1 = KW;

  float acc = 0.0f;
  for (int t = t0; t < t1; t++) {
    const float* xr = xb + (size_t)(l + t - 64) * DIM;
#pragma unroll 8
    for (int c = 0; c < CPG; c++)
      acc += wd[c * KW + t] * xr[c];
  }
  out[idx] = x[idx] + geluf(acc + cb[d]);
}

// ---------------------------------------------------------------------------
// LayerNorm over D=768; one 256-thread block per row. Writes f32 and a bf16
// copy (the bf16 copy feeds the WMMA GEMMs directly).
// ---------------------------------------------------------------------------
__global__ __launch_bounds__(256) void layernorm_kernel(
    const float* __restrict__ in, const float* __restrict__ gamma,
    const float* __restrict__ beta, float* __restrict__ outF,
    __bf16* __restrict__ outB) {
  __shared__ float red[256];
  const int row = blockIdx.x, tid = threadIdx.x;
  const float* p = in + (size_t)row * DIM;
  float x0 = p[tid], x1 = p[tid + 256], x2 = p[tid + 512];

  red[tid] = x0 + x1 + x2;
  __syncthreads();
  for (int o = 128; o > 0; o >>= 1) {
    if (tid < o) red[tid] += red[tid + o];
    __syncthreads();
  }
  float mean = red[0] * (1.0f / DIM);
  __syncthreads();

  float d0 = x0 - mean, d1 = x1 - mean, d2 = x2 - mean;
  red[tid] = d0 * d0 + d1 * d1 + d2 * d2;
  __syncthreads();
  for (int o = 128; o > 0; o >>= 1) {
    if (tid < o) red[tid] += red[tid + o];
    __syncthreads();
  }
  float rs = rsqrtf(red[0] * (1.0f / DIM) + 1e-5f);

  float y0 = d0 * rs * gamma[tid]       + beta[tid];
  float y1 = d1 * rs * gamma[tid + 256] + beta[tid + 256];
  float y2 = d2 * rs * gamma[tid + 512] + beta[tid + 512];
  float* q = outF + (size_t)row * DIM;
  q[tid] = y0; q[tid + 256] = y1; q[tid + 512] = y2;
  __bf16* qb = outB + (size_t)row * DIM;
  qb[tid] = (__bf16)y0; qb[tid + 256] = (__bf16)y1; qb[tid + 512] = (__bf16)y2;
}

// ---------------------------------------------------------------------------
// bf16 WMMA GEMM:  out(MxN f32) = A(MxK bf16) @ W(KxN bf16) + bias
//                  [GELU] [+residual f32]; f32 and/or bf16 outputs are
//                  compile-time selected (no epilogue branches).
// Block: 256 thr = 8 waves; block tile 128x128; wave tile 32x64; K-step 32.
// A staged row-major in LDS, W staged transposed ([n][k]) so both A and B
// WMMA fragments are contiguous 16B LDS reads per the documented layouts.
// Software pipelined: next K-tile's global loads issue before the current
// tile's WMMAs, so HBM/L2 latency hides behind matrix math (split counters
// keep the loads outstanding across the compute phase).
// ---------------------------------------------------------------------------
template <bool GELU, bool RES, bool OUTF, bool OUTB>
__global__ __launch_bounds__(256) void gemm_bf16_kernel(
    const __bf16* __restrict__ A, const __bf16* __restrict__ W,
    const float* __restrict__ bias, const float* __restrict__ residual,
    float* __restrict__ outF, __bf16* __restrict__ outB, int N, int Kd) {
  __shared__ __bf16 ldsA[128][40];   // [m][k]  (+8 pad)
  __shared__ __bf16 ldsBt[128][40];  // [n][k]  (+8 pad)

  const int tid = threadIdx.x;
  const int lane = tid & 31, wave = tid >> 5;
  const int wm = wave >> 1, wn = wave & 1;       // wave grid 4(M) x 2(N)
  const int r = lane & 15, hi = lane >> 4;
  const int mBase = blockIdx.y * 128;
  const int nBase = blockIdx.x * 128;

  v8f acc[2][4];
  const v8f vz = {0, 0, 0, 0, 0, 0, 0, 0};
#pragma unroll
  for (int i = 0; i < 2; i++)
#pragma unroll
    for (int j = 0; j < 4; j++) acc[i][j] = vz;

  const int aRow = tid >> 1, aCol = (tid & 1) * 16;   // 128 rows x 32 cols
  const int bK = tid >> 3, bN = (tid & 7) * 16;       // 32 rows x 128 cols
  const __bf16* aPtr = A + (size_t)(mBase + aRow) * Kd + aCol;
  const __bf16* wPtr = W + (size_t)bK * N + nBase + bN;
  const size_t wStep = (size_t)32 * N;

  // pipeline prologue: fetch tile 0
  v8bf a0 = *(const v8bf*)aPtr;
  v8bf a1 = *(const v8bf*)(aPtr + 8);
  v8bf w0 = *(const v8bf*)wPtr;
  v8bf w1 = *(const v8bf*)(wPtr + 8);

  for (int kt = 0; kt < Kd; kt += 32) {
    // commit prefetched tile to LDS
    *(v8bf*)&ldsA[aRow][aCol] = a0;
    *(v8bf*)&ldsA[aRow][aCol + 8] = a1;
#pragma unroll
    for (int i = 0; i < 8; i++) ldsBt[bN + i][bK] = w0[i];
#pragma unroll
    for (int i = 0; i < 8; i++) ldsBt[bN + 8 + i][bK] = w1[i];
    __syncthreads();

    // issue next tile's global loads early (latency hidden by WMMAs below)
    if (kt + 32 < Kd) {
      const __bf16* an = aPtr + kt + 32;
      const __bf16* wn_ = wPtr + (size_t)(kt + 32) * N;
      a0 = *(const v8bf*)an;
      a1 = *(const v8bf*)(an + 8);
      w0 = *(const v8bf*)wn_;
      w1 = *(const v8bf*)(wn_ + 8);
    }

    v16bf af[2];
#pragma unroll
    for (int ms = 0; ms < 2; ms++) {
      int row = wm * 32 + ms * 16 + r;
      v8bf lo = *(const v8bf*)&ldsA[row][hi * 8];
      v8bf hh = *(const v8bf*)&ldsA[row][16 + hi * 8];
      af[ms] = cat16(lo, hh);
    }
#pragma unroll
    for (int t = 0; t < 4; t++) {
      int col = wn * 64 + t * 16 + r;
      v8bf lo = *(const v8bf*)&ldsBt[col][hi * 8];
      v8bf hh = *(const v8bf*)&ldsBt[col][16 + hi * 8];
      v16bf bfr = cat16(lo, hh);
#pragma unroll
      for (int ms = 0; ms < 2; ms++)
        acc[ms][t] = wmma_bf(af[ms], bfr, acc[ms][t]);
    }
    __syncthreads();
  }

  // Epilogue. C layout: lane(0-15)=col r, VGPR j = row j (+8 for hi half).
#pragma unroll
  for (int ms = 0; ms < 2; ms++) {
#pragma unroll
    for (int t = 0; t < 4; t++) {
      int col = nBase + wn * 64 + t * 16 + r;
      float bv = bias[col];
#pragma unroll
      for (int j = 0; j < 8; j++) {
        int row = mBase + wm * 32 + ms * 16 + hi * 8 + j;
        float v = acc[ms][t][j] + bv;
        if (GELU) v = geluf(v);
        size_t idx = (size_t)row * N + col;
        if (RES) v += residual[idx];
        if (OUTF) outF[idx] = v;
        if (OUTB) outB[idx] = (__bf16)v;
      }
    }
  }
}

// ---------------------------------------------------------------------------
// Flash-style attention, bf16 WMMA.  Q/K/V layout: (B, L, H*HDIM) bf16.
// Grid (L/128, H, B); 8 waves; each wave owns 16 queries x all 64 head dims.
// Never materializes the LxL score matrix; online softmax in C-layout regs
// with shfl_xor row-reductions within each 16-lane half.
// ---------------------------------------------------------------------------
__global__ __launch_bounds__(256) void attn_kernel(
    const __bf16* __restrict__ Q, const __bf16* __restrict__ Kv,
    const __bf16* __restrict__ Vv, __bf16* __restrict__ O) {
  __shared__ __bf16 ldsK[64][72];    // [key][hd]
  __shared__ __bf16 ldsVt[64][72];   // [hd][key]
  __shared__ __bf16 ldsP[128][72];   // per-wave rows [wave*16 .. +15]

  const int tid = threadIdx.x, lane = tid & 31, wave = tid >> 5;
  const int r = lane & 15, hi = lane >> 4;
  const int qBlock = blockIdx.x * 128;
  const int h = blockIdx.y, b = blockIdx.z;
  const size_t headOff = (size_t)h * HDIM;

  // Q fragments for this wave's 16 query rows (A layout, 2 x 32-wide k-steps)
  v16bf qf[2];
  {
    size_t qrow = (size_t)b * SEQ + qBlock + wave * 16 + r;
    const __bf16* qp = Q + qrow * DIM + headOff;
#pragma unroll
    for (int ks = 0; ks < 2; ks++) {
      v8bf lo = *(const v8bf*)(qp + ks * 32 + hi * 8);
      v8bf hh = *(const v8bf*)(qp + ks * 32 + 16 + hi * 8);
      qf[ks] = cat16(lo, hh);
    }
  }

  float rowM[8], rowS[8];
  v8f accO[4];
  const v8f vz = {0, 0, 0, 0, 0, 0, 0, 0};
#pragma unroll
  for (int j = 0; j < 8; j++) { rowM[j] = -1e30f; rowS[j] = 0.0f; }
#pragma unroll
  for (int t = 0; t < 4; t++) accO[t] = vz;

  const int sRow = tid >> 2, sC0 = (tid & 3) * 16;   // 64 x 64 staging

  for (int kb = 0; kb < SEQ; kb += 64) {
    { // stage K (row-major) and V (transposed)
      size_t krow = (size_t)b * SEQ + kb + sRow;
      const __bf16* kp = Kv + krow * DIM + headOff + sC0;
      v8bf k0 = *(const v8bf*)kp;
      v8bf k1 = *(const v8bf*)(kp + 8);
      *(v8bf*)&ldsK[sRow][sC0] = k0;
      *(v8bf*)&ldsK[sRow][sC0 + 8] = k1;
      const __bf16* vp = Vv + krow * DIM + headOff + sC0;
      v8bf v0 = *(const v8bf*)vp;
      v8bf v1 = *(const v8bf*)(vp + 8);
#pragma unroll
      for (int i = 0; i < 8; i++) ldsVt[sC0 + i][sRow] = v0[i];
#pragma unroll
      for (int i = 0; i < 8; i++) ldsVt[sC0 + 8 + i][sRow] = v1[i];
    }
    __syncthreads();

    // S = Q @ K^T  (4 key sub-tiles x 2 hd k-steps)
    v8f sacc[4];
#pragma unroll
    for (int t = 0; t < 4; t++) sacc[t] = vz;
#pragma unroll
    for (int t = 0; t < 4; t++) {
      int key = t * 16 + r;
#pragma unroll
      for (int ks = 0; ks < 2; ks++) {
        v8bf lo = *(const v8bf*)&ldsK[key][ks * 32 + hi * 8];
        v8bf hh = *(const v8bf*)&ldsK[key][ks * 32 + 16 + hi * 8];
        sacc[t] = wmma_bf(qf[ks], cat16(lo, hh), sacc[t]);
      }
    }

    // online softmax (row reductions within each 16-lane half; rows of the
    // C layout live at VGPR j for lanes 0-15 and j+8 for lanes 16-31)
    float mloc[8];
#pragma unroll
    for (int j = 0; j < 8; j++) mloc[j] = sacc[0][j];
#pragma unroll
    for (int t = 1; t < 4; t++)
#pragma unroll
      for (int j = 0; j < 8; j++) mloc[j] = fmaxf(mloc[j], sacc[t][j]);
#pragma unroll
    for (int m = 1; m < 16; m <<= 1)
#pragma unroll
      for (int j = 0; j < 8; j++)
        mloc[j] = fmaxf(mloc[j], __shfl_xor(mloc[j], m));

    float newM[8], resc[8], suml[8];
#pragma unroll
    for (int j = 0; j < 8; j++) {
      newM[j] = fmaxf(rowM[j], mloc[j] * ATT_SCALE);
      resc[j] = __expf(rowM[j] - newM[j]);
      suml[j] = 0.0f;
    }
#pragma unroll
    for (int t = 0; t < 4; t++)
#pragma unroll
      for (int j = 0; j < 8; j++) {
        float p = __expf(sacc[t][j] * ATT_SCALE - newM[j]);
        suml[j] += p;
        ldsP[wave * 16 + hi * 8 + j][t * 16 + r] = (__bf16)p;
      }
#pragma unroll
    for (int m = 1; m < 16; m <<= 1)
#pragma unroll
      for (int j = 0; j < 8; j++) suml[j] += __shfl_xor(suml[j], m);
#pragma unroll
    for (int j = 0; j < 8; j++) {
      rowS[j] = rowS[j] * resc[j] + suml[j];
      rowM[j] = newM[j];
    }
#pragma unroll
    for (int t = 0; t < 4; t++)
#pragma unroll
      for (int j = 0; j < 8; j++) accO[t][j] = accO[t][j] * resc[j];

    // O += P @ V  (P re-read from own-wave LDS in A layout; same-wave DScnt
    // ordering is handled by compiler-inserted s_wait_dscnt)
#pragma unroll
    for (int ks = 0; ks < 2; ks++) {
      const __bf16* pp = &ldsP[wave * 16 + r][0];
      v8bf lo = *(const v8bf*)(pp + ks * 32 + hi * 8);
      v8bf hh = *(const v8bf*)(pp + ks * 32 + 16 + hi * 8);
      v16bf pf = cat16(lo, hh);
#pragma unroll
      for (int t = 0; t < 4; t++) {
        int hd = t * 16 + r;
        v8bf vlo = *(const v8bf*)&ldsVt[hd][ks * 32 + hi * 8];
        v8bf vhh = *(const v8bf*)&ldsVt[hd][ks * 32 + 16 + hi * 8];
        accO[t] = wmma_bf(pf, cat16(vlo, vhh), accO[t]);
      }
    }
    __syncthreads();
  }

  // write O (bf16) back in (B, L, H*HDIM)
#pragma unroll
  for (int t = 0; t < 4; t++) {
    int hd = t * 16 + r;
#pragma unroll
    for (int j = 0; j < 8; j++) {
      int qrow = qBlock + wave * 16 + hi * 8 + j;
      float v = accO[t][j] / rowS[j];
      O[((size_t)b * SEQ + qrow) * DIM + headOff + hd] = (__bf16)v;
    }
  }
}

// ---------------------------------------------------------------------------
// Host-side orchestration
// ---------------------------------------------------------------------------
extern "C" void kernel_launch(void* const* d_in, const int* in_sizes, int n_in,
                              void* d_out, int out_size, void* d_ws,
                              size_t ws_size, hipStream_t stream) {
  (void)in_sizes; (void)n_in; (void)out_size; (void)ws_size;
  const float* x     = (const float*)d_in[0];
  const float* convw = (const float*)d_in[1];
  const float* convb = (const float*)d_in[2];
  const float* ln0g  = (const float*)d_in[3];
  const float* ln0b  = (const float*)d_in[4];
  const float* Wq    = (const float*)d_in[5];
  const float* bq    = (const float*)d_in[6];
  const float* Wk    = (const float*)d_in[7];
  const float* bk    = (const float*)d_in[8];
  const float* Wv    = (const float*)d_in[9];
  const float* bv    = (const float*)d_in[10];
  const float* Wo    = (const float*)d_in[11];
  const float* bo    = (const float*)d_in[12];
  const float* ln1g  = (const float*)d_in[13];
  const float* ln1b  = (const float*)d_in[14];
  const float* W1    = (const float*)d_in[15];
  const float* b1    = (const float*)d_in[16];
  const float* W2    = (const float*)d_in[17];
  const float* b2    = (const float*)d_in[18];
  const float* ln2g  = (const float*)d_in[19];
  const float* ln2b  = (const float*)d_in[20];

  char* ws = (char*)d_ws;
  size_t off = 0;
  auto alloc = [&](size_t bytes) -> void* {
    void* p = ws + off;
    off += (bytes + 255) & ~(size_t)255;
    return p;
  };
  float*  hf  = (float*) alloc(sizeof(float)  * MROWS * DIM);  // hidden (f32)
  float*  sf  = (float*) alloc(sizeof(float)  * MROWS * DIM);  // pre-LN sum
  __bf16* hb  = (__bf16*)alloc(sizeof(__bf16) * MROWS * DIM);  // hidden (bf16)
  __bf16* qb  = (__bf16*)alloc(sizeof(__bf16) * MROWS * DIM);
  __bf16* kbf = (__bf16*)alloc(sizeof(__bf16) * MROWS * DIM);
  __bf16* vbf = (__bf16*)alloc(sizeof(__bf16) * MROWS * DIM);
  __bf16* ob  = (__bf16*)alloc(sizeof(__bf16) * MROWS * DIM);
  __bf16* mb  = (__bf16*)alloc(sizeof(__bf16) * MROWS * FF);   // FFN mid
  __bf16* wqb = (__bf16*)alloc(sizeof(__bf16) * DIM * DIM);
  __bf16* wkb = (__bf16*)alloc(sizeof(__bf16) * DIM * DIM);
  __bf16* wvb = (__bf16*)alloc(sizeof(__bf16) * DIM * DIM);
  __bf16* wob = (__bf16*)alloc(sizeof(__bf16) * DIM * DIM);
  __bf16* w1b = (__bf16*)alloc(sizeof(__bf16) * DIM * FF);
  __bf16* w2b = (__bf16*)alloc(sizeof(__bf16) * FF * DIM);

  dim3 blk(256);

  // positional conv + GELU + residual, then LN0 (writes f32 + bf16)
  conv_pos_kernel<<<(MROWS * DIM) / 256, blk, 0, stream>>>(x, convw, convb, hf);
  layernorm_kernel<<<MROWS, blk, 0, stream>>>(hf, ln0g, ln0b, hf, hb);

  const int ndd = DIM * DIM, ndf = DIM * FF;
  dim3 g768(DIM / 128, MROWS / 128);    // (6, 32)
  dim3 g3072(FF / 128, MROWS / 128);    // (24, 32)
  dim3 ga(SEQ / 128, NH, NB);           // (16, 12, 2)

  for (int l = 0; l < NLAYER; l++) {
    // weight conversion to bf16 (read once, tiny vs GEMM cost)
    cvt_bf16_kernel<<<(ndd + 255) / 256, blk, 0, stream>>>(Wq + (size_t)l * ndd, wqb, ndd);
    cvt_bf16_kernel<<<(ndd + 255) / 256, blk, 0, stream>>>(Wk + (size_t)l * ndd, wkb, ndd);
    cvt_bf16_kernel<<<(ndd + 255) / 256, blk, 0, stream>>>(Wv + (size_t)l * ndd, wvb, ndd);
    cvt_bf16_kernel<<<(ndd + 255) / 256, blk, 0, stream>>>(Wo + (size_t)l * ndd, wob, ndd);
    cvt_bf16_kernel<<<(ndf + 255) / 256, blk, 0, stream>>>(W1 + (size_t)l * ndf, w1b, ndf);
    cvt_bf16_kernel<<<(ndf + 255) / 256, blk, 0, stream>>>(W2 + (size_t)l * ndf, w2b, ndf);

    // Q, K, V projections (bf16 out feeds attention directly)
    gemm_bf16_kernel<false, false, false, true><<<g768, blk, 0, stream>>>(
        hb, wqb, bq + (size_t)l * DIM, nullptr, nullptr, qb, DIM, DIM);
    gemm_bf16_kernel<false, false, false, true><<<g768, blk, 0, stream>>>(
        hb, wkb, bk + (size_t)l * DIM, nullptr, nullptr, kbf, DIM, DIM);
    gemm_bf16_kernel<false, false, false, true><<<g768, blk, 0, stream>>>(
        hb, wvb, bv + (size_t)l * DIM, nullptr, nullptr, vbf, DIM, DIM);

    // attention
    attn_kernel<<<ga, blk, 0, stream>>>(qb, kbf, vbf, ob);

    // output projection + residual, LN1
    gemm_bf16_kernel<false, true, true, false><<<g768, blk, 0, stream>>>(
        ob, wob, bo + (size_t)l * DIM, hf, sf, nullptr, DIM, DIM);
    layernorm_kernel<<<MROWS, blk, 0, stream>>>(sf, ln1g + (size_t)l * DIM,
        ln1b + (size_t)l * DIM, hf, hb);

    // FFN: gelu(h @ W1 + b1) @ W2 + b2 + residual, LN2
    gemm_bf16_kernel<true, false, false, true><<<g3072, blk, 0, stream>>>(
        hb, w1b, b1 + (size_t)l * FF, nullptr, nullptr, mb, FF, DIM);
    gemm_bf16_kernel<false, true, true, false><<<g768, blk, 0, stream>>>(
        mb, w2b, b2 + (size_t)l * DIM, hf, sf, nullptr, DIM, FF);
    layernorm_kernel<<<MROWS, blk, 0, stream>>>(sf, ln2g + (size_t)l * DIM,
        ln2b + (size_t)l * DIM, hf, hb);
  }

  hipMemcpyAsync(d_out, hf, sizeof(float) * MROWS * DIM,
                 hipMemcpyDeviceToDevice, stream);
}
